// QuantumClassicalHybridModel_69226282877371
// MI455X (gfx1250) — matrix operand order repaired and checked by reading, not compiled, not run on hardware
//
#include <hip/hip_runtime.h>
#include <hip/hip_bf16.h>

typedef __attribute__((ext_vector_type(16))) _Float16 v16h;
typedef __attribute__((ext_vector_type(8)))  float    v8f;

#define PI_F 3.14159265358979323846f

// =====================================================================
// Quantum circuit simulation: one wave32 per sample.
// 256 amplitudes = 32 lanes x 8 register slots.
// amp index i[7:0]: bits 7..3 = lane bits 4..0 (wires 0..4, MSB-first),
//                   bits 2..0 = slot bits (wires 5..7).
// =====================================================================

__device__ __forceinline__ void ry_lane(float re[8], float im[8], int m,
                                        float c, float s, int lane) {
  const float sgn = (lane & m) ? s : -s;
#pragma unroll
  for (int k = 0; k < 8; ++k) {
    float pr = __shfl_xor(re[k], m, 32);
    float pi = __shfl_xor(im[k], m, 32);
    re[k] = c * re[k] + sgn * pr;
    im[k] = c * im[k] + sgn * pi;
  }
}

__device__ __forceinline__ void ry_slot(float re[8], float im[8], int sm,
                                        float c, float s) {
#pragma unroll
  for (int k = 0; k < 8; ++k) {
    if (!(k & sm)) {
      const int k1 = k | sm;
      float r0 = re[k], r1 = re[k1];
      re[k]  = c * r0 - s * r1;
      re[k1] = s * r0 + c * r1;
      float i0 = im[k], i1 = im[k1];
      im[k]  = c * i0 - s * i1;
      im[k1] = s * i0 + c * i1;
    }
  }
}

__device__ __forceinline__ void ry_gate(float re[8], float im[8], int w,
                                        float c, float s, int lane) {
  if (w < 5) ry_lane(re, im, 1 << (4 - w), c, s, lane);
  else       ry_slot(re, im, 1 << (7 - w), c, s);
}

// RZ: diagonal phase, purely lane-local.
__device__ __forceinline__ void rz_gate(float re[8], float im[8], int w,
                                        float c, float s, int lane) {
#pragma unroll
  for (int k = 0; k < 8; ++k) {
    const int bit = (w < 5) ? ((lane >> (4 - w)) & 1) : ((k >> (7 - w)) & 1);
    const float sv = bit ? s : -s;
    float r = re[k], q = im[k];
    re[k] = c * r - sv * q;
    im[k] = c * q + sv * r;
  }
}

__device__ __forceinline__ void cnot_ll(float re[8], float im[8], int mc, int mt, int lane) {
  const bool ctl = (lane & mc) != 0;
#pragma unroll
  for (int k = 0; k < 8; ++k) {
    float pr = __shfl_xor(re[k], mt, 32);
    float pi = __shfl_xor(im[k], mt, 32);
    re[k] = ctl ? pr : re[k];
    im[k] = ctl ? pi : im[k];
  }
}

__device__ __forceinline__ void cnot_ls(float re[8], float im[8], int mc, int st, int lane) {
  const bool ctl = (lane & mc) != 0;
#pragma unroll
  for (int k = 0; k < 8; ++k) {
    if (!(k & st)) {
      const int k1 = k | st;
      float r0 = re[k], r1 = re[k1];
      re[k]  = ctl ? r1 : r0;
      re[k1] = ctl ? r0 : r1;
      float i0 = im[k], i1 = im[k1];
      im[k]  = ctl ? i1 : i0;
      im[k1] = ctl ? i0 : i1;
    }
  }
}

__device__ __forceinline__ void cnot_sl(float re[8], float im[8], int sc, int mt) {
#pragma unroll
  for (int k = 0; k < 8; ++k) {
    if (k & sc) {  // compile-time under unroll
      re[k] = __shfl_xor(re[k], mt, 32);
      im[k] = __shfl_xor(im[k], mt, 32);
    }
  }
}

__device__ __forceinline__ void cnot_ss(float re[8], float im[8], int sc, int st) {
#pragma unroll
  for (int k = 0; k < 8; ++k) {
    if ((k & sc) && !(k & st)) {
      const int k1 = k | st;
      float t = re[k]; re[k] = re[k1]; re[k1] = t;
      t = im[k]; im[k] = im[k1]; im[k1] = t;
    }
  }
}

__global__ __launch_bounds__(256) void qsim_kernel(const float* __restrict__ x,
                                                   const float* __restrict__ qw,
                                                   float* __restrict__ qout, int B) {
  const int lane = threadIdx.x & 31;
  const int s = blockIdx.x * 8 + (threadIdx.x >> 5);
  if (s >= B) return;

  // per-sample min-max normalization -> RY encoding angles in [0, pi]
  float ang[8];
  float xmn = 3.4e38f, xmx = -3.4e38f;
#pragma unroll
  for (int w = 0; w < 8; ++w) {
    float v = x[(size_t)s * 8 + w];
    ang[w] = v;
    xmn = fminf(xmn, v);
    xmx = fmaxf(xmx, v);
  }
  const float sc = PI_F / (xmx - xmn + 1e-8f);

  float re[8], im[8];
#pragma unroll
  for (int k = 0; k < 8; ++k) { re[k] = 0.f; im[k] = 0.f; }
  if (lane == 0) re[0] = 1.f;  // |00000000>

  // input encoding
#pragma unroll
  for (int w = 0; w < 8; ++w) {
    float th = 0.5f * (ang[w] - xmn) * sc;
    float ss, cc;
    __sincosf(th, &ss, &cc);
    ry_gate(re, im, w, cc, ss, lane);
  }

  // variational layers
#pragma unroll
  for (int l = 0; l < 3; ++l) {
#pragma unroll
    for (int w = 0; w < 8; ++w) {
      float ss, cc;
      __sincosf(0.5f * qw[(l * 8 + w) * 2 + 0], &ss, &cc);
      ry_gate(re, im, w, cc, ss, lane);
      __sincosf(0.5f * qw[(l * 8 + w) * 2 + 1], &ss, &cc);
      rz_gate(re, im, w, cc, ss, lane);
    }
    cnot_ll(re, im, 16, 8, lane);
    cnot_ll(re, im, 8, 4, lane);
    cnot_ll(re, im, 4, 2, lane);
    cnot_ll(re, im, 2, 1, lane);
    cnot_ls(re, im, 1, 4, lane);
    cnot_ss(re, im, 4, 2);
    cnot_ss(re, im, 2, 1);
    cnot_sl(re, im, 1, 16);
  }

  // <Z_w> = sum_i (-1)^{bit_w(i)} |amp_i|^2
  float pk[8], psum = 0.f;
#pragma unroll
  for (int k = 0; k < 8; ++k) {
    pk[k] = re[k] * re[k] + im[k] * im[k];
    psum += pk[k];
  }
  float ez[8];
#pragma unroll
  for (int w = 0; w < 5; ++w)
    ez[w] = ((lane >> (4 - w)) & 1) ? -psum : psum;
#pragma unroll
  for (int w = 5; w < 8; ++w) {
    float t = 0.f;
#pragma unroll
    for (int k = 0; k < 8; ++k)
      t += ((k >> (7 - w)) & 1) ? -pk[k] : pk[k];
    ez[w] = t;
  }
#pragma unroll
  for (int off = 16; off > 0; off >>= 1) {
#pragma unroll
    for (int w = 0; w < 8; ++w) ez[w] += __shfl_xor(ez[w], off, 32);
  }
  if (lane == 0) {
#pragma unroll
    for (int w = 0; w < 8; ++w) qout[(size_t)s * 8 + w] = ez[w];
  }
}

// =====================================================================
// Fused MLP with v_wmma_f32_16x16x32_f16.
// Block = 256 threads = 8 waves; each wave owns a 16-sample tile.
//   GEMM1: (16x8 pad 32) x (32x64)  -> 4 WMMAs, +b1, ReLU
//   GEMM2: (16x64)       x (64x256) -> 32 WMMAs, +b2
//
// Fragment layouts (CDNA5 ISA 7.12.2, wave32), VGPR v == word u[v]:
//   A 16x32 f16: M = lane&15; word v = K-pair (kb, kb+1),
//                kb = (v&4?16:0) + 8*(lane>>4) + 2*(v&3)
//   B 32x16 f16: N = lane&15; word v = K-pair (kb, kb+1), kb as for A (W1 staged as 16*(lane>>4)+2v)
//   C/D 16x16 f32: VGPR r -> M = r + 8*(lane>>4), N = lane&15
//
// W1/W2 are re-staged in LDS in FRAGMENT-NATIVE order:
//   wXf[frag][part(0/1)][lane][4 packed f16-pair words]
// so every B-fragment is exactly two ds_load_b128 straight into the
// contiguous VGPR octet (lane stride = 4 words -> each 16-lane phase
// covers all 64 banks once, conflict-free, zero register shuffling).
// The h tile uses stride 72 halves (36 words): its A2 fragment groups
// (nl*36 + 16kc + 4lh + v) and (+8) are 16B-aligned contiguous runs,
// also two ds_load_b128 each.
// =====================================================================

#define HS 72  // halves per h-tile row; 36 words, 16B-aligned fragment runs

union frag16 { unsigned u[8]; v16h v; };

__device__ __forceinline__ unsigned pack2f16(float a, float b) {
  union { _Float16 h[2]; unsigned u; } t;
  t.h[0] = (_Float16)a;
  t.h[1] = (_Float16)b;
  return t.u;
}

__global__ __launch_bounds__(256) void mlp_wmma_kernel(
    const float* __restrict__ q, const float* __restrict__ W1,
    const float* __restrict__ b1, const float* __restrict__ W2,
    const float* __restrict__ b2, float* __restrict__ out) {
  // fragment-native weight copies (packed f16-pair words)
  __shared__ __align__(16) unsigned w1f[4 * 256];    //  4 KB: [n][part][lane][4]
  __shared__ __align__(16) unsigned w2f[32 * 256];   // 32 KB: [j*2+kc][part][lane][4]
  __shared__ __align__(16) _Float16 hs[8][16 * HS];  // 18 KB: per-wave 16x64 h tile

  const int tid  = threadIdx.x;
  const int lane = tid & 31;
  const int wv   = tid >> 5;
  const int lh   = lane >> 4;   // lane half
  const int nl   = lane & 15;

  // ---- stage W1 (zero-padded to K=32) in fragment-native order ----
  for (int i = tid; i < 4 * 256; i += 256) {
    const int vi = i & 3, ln = (i >> 2) & 31, part = (i >> 7) & 1, n = i >> 8;
    const int v = part * 4 + vi;
    const int K = 16 * (ln >> 4) + 2 * v;       // K-pair base for word v
    const int N = 16 * n + (ln & 15);
    unsigned w = 0u;
    if (K < 8) w = pack2f16(W1[K * 64 + N], W1[(K + 1) * 64 + N]);
    w1f[i] = w;
  }
  // ---- stage W2 in fragment-native order ----
  for (int i = tid; i < 32 * 256; i += 256) {
    const int vi = i & 3, ln = (i >> 2) & 31, part = (i >> 7) & 1, frag = i >> 8;
    const int j = frag >> 1, kc = frag & 1;
    const int v = part * 4 + vi;
    const int K = 32 * kc + ((v & 4) ? 16 : 0) + 8 * (ln >> 4) + 2 * (v & 3);
    const int N = 16 * j + (ln & 15);
    w2f[i] = pack2f16(W2[K * 256 + N], W2[(K + 1) * 256 + N]);
  }
  __syncthreads();

  const int s0 = (blockIdx.x * 8 + wv) * 16;  // first sample of this tile

  // ---- A fragment for GEMM1: unconditional global loads + cndmask select ----
  float qv[8];
#pragma unroll
  for (int k = 0; k < 8; ++k) qv[k] = q[(size_t)(s0 + nl) * 8 + k];

  frag16 a1;
#pragma unroll
  for (int v = 0; v < 4; ++v)          // lh==0: K = 2v,2v+1 (real); lh==1: K>=8 -> 0
    a1.u[v] = lh ? 0u : pack2f16(qv[2 * v], qv[2 * v + 1]);
#pragma unroll
  for (int v = 4; v < 8; ++v) a1.u[v] = 0u;   // K >= 16 -> 0

  // ---- GEMM1: 4 WMMAs over the 64 hidden cols ----
  v8f hacc[4];
#pragma unroll
  for (int n = 0; n < 4; ++n) {
    frag16 bf;
    *(uint4*)&bf.u[0] = *(const uint4*)&w1f[(n * 2 + 0) * 128 + lane * 4];
    *(uint4*)&bf.u[4] = *(const uint4*)&w1f[(n * 2 + 1) * 128 + lane * 4];
    v8f z = {};
    hacc[n] = __builtin_amdgcn_wmma_f32_16x16x32_f16(false, a1.v, false, bf.v,
                                                     (short)0, z, false, false);
  }

  // ---- bias + ReLU; park h tile (16x64 f16) in this wave's LDS slab ----
#pragma unroll
  for (int n = 0; n < 4; ++n) {
#pragma unroll
    for (int r = 0; r < 8; ++r) {
      const int M = r + 8 * lh;
      const int N = 16 * n + nl;
      float v = hacc[n][r] + b1[N];
      hs[wv][M * HS + N] = (_Float16)fmaxf(v, 0.f);
    }
  }
  // wave-private LDS slab: same-wave DS ordering suffices, no barrier needed

  // ---- reload h as two 16x32 A fragments (2x ds_load_b128 each) ----
  const unsigned* hw = (const unsigned*)&hs[wv][0];
  frag16 a2[2];
#pragma unroll
  for (int kc = 0; kc < 2; ++kc) {
    const int base = nl * (HS / 2) + 16 * kc + 4 * lh;  // word index, 16B aligned
    *(uint4*)&a2[kc].u[0] = *(const uint4*)&hw[base];
    *(uint4*)&a2[kc].u[4] = *(const uint4*)&hw[base + 8];
  }

  // ---- GEMM2: 16 N-tiles x 2 K-chunks = 32 WMMAs, fused bias + store ----
#pragma unroll
  for (int j = 0; j < 16; ++j) {
    v8f d = {};
#pragma unroll
    for (int kc = 0; kc < 2; ++kc) {
      const int frag = j * 2 + kc;
      frag16 bf;
      *(uint4*)&bf.u[0] = *(const uint4*)&w2f[(frag * 2 + 0) * 128 + lane * 4];
      *(uint4*)&bf.u[4] = *(const uint4*)&w2f[(frag * 2 + 1) * 128 + lane * 4];
      d = __builtin_amdgcn_wmma_f32_16x16x32_f16(false, a2[kc].v, false, bf.v,
                                                 (short)0, d, false, false);
    }
    const float bb = b2[16 * j + nl];
#pragma unroll
    for (int r = 0; r < 8; ++r) {
      const int M = r + 8 * lh;
      out[(size_t)(s0 + M) * 256 + 16 * j + nl] = d[r] + bb;
    }
  }
}

// =====================================================================
extern "C" void kernel_launch(void* const* d_in, const int* in_sizes, int n_in,
                              void* d_out, int out_size, void* d_ws, size_t ws_size,
                              hipStream_t stream) {
  (void)n_in; (void)out_size; (void)ws_size;
  const float* x  = (const float*)d_in[0];
  const float* qw = (const float*)d_in[1];
  const float* W1 = (const float*)d_in[2];
  const float* b1 = (const float*)d_in[3];
  const float* W2 = (const float*)d_in[4];
  const float* b2 = (const float*)d_in[5];
  float* out = (float*)d_out;

  const int B = in_sizes[0] / 8;          // 8192 samples
  float* qbuf = (float*)d_ws;             // B x 8 floats of scratch

  const int blocksA = (B + 7) / 8;
  qsim_kernel<<<blocksA, 256, 0, stream>>>(x, qw, qbuf, B);

  const int blocksB = B / 128;            // 8 waves x 16-sample tiles per block
  mlp_wmma_kernel<<<blocksB, 256, 0, stream>>>(qbuf, W1, b1, W2, b2, out);
}
